// Decoder_60112362275056
// MI455X (gfx1250) — matrix operand order repaired and checked by reading, not compiled, not run on hardware
//
#include <hip/hip_runtime.h>
#include <hip/hip_bf16.h>

// Persistent GRU decoder for MI455X (gfx1250, wave32).
//   B=128, T=512, X=64, H=256.
//   Grid = 12 workgroups x 256 threads (8 waves):
//     WG 0..7  : 32 H-columns each -> 96 gate columns (r,z,n), WMMA bf16 gemms
//                from LDS-resident weight slices, then elementwise gates.
//     WG 8..11 : 16 output columns each of the (h_prev + skip) @ W_out.T gemm,
//                with W_out fragments held in VGPRs across all 512 steps.
//   Fragments are loaded as 2x ds_load_b128 each (per-lane 16B runs are
//   contiguous in our LDS layouts). A-fragments are hoisted across the 6
//   column tiles (kt-outer loop, 8 live f32 accumulators).
//   Per-step grid sync via epoch spin barrier in workspace (12 resident blocks).
//   Workspace: [bar(256B)][h ping-pong 2*B*H f32][buf T*B*H f32 (~64MB, L2-resident)].

typedef __attribute__((ext_vector_type(16))) __bf16 v16bf;
typedef __attribute__((ext_vector_type(8)))  float  v8f;

#define B_   128
#define T_   512
#define X_   64
#define H_   256
#define NGATE_WG 8
#define NOUT_WG  4
#define NWG (NGATE_WG + NOUT_WG)
#define NTHREADS 256

__device__ __forceinline__ unsigned short f2bfu(float f) {
  union { float f; unsigned u; } v; v.f = f;
  unsigned r = v.u + 0x7FFFu + ((v.u >> 16) & 1u);   // round-to-nearest-even
  return (unsigned short)(r >> 16);
}
__device__ __forceinline__ unsigned pack2bf(float a, float b) {
  return (unsigned)f2bfu(a) | ((unsigned)f2bfu(b) << 16);
}

__device__ __forceinline__ float fsigmoid(float x) {
  return 1.0f / (1.0f + __expf(-x));
}
__device__ __forceinline__ float ftanh_fast(float x) {
  return 1.0f - 2.0f / (__expf(2.0f * x) + 1.0f);
}

union FragU { v16bf v; uint4 q[2]; };

// A-matrix 16x32 bf16 fragment (ISA 7.12.2): lanes 0-15 hold row M=lane,
// K in {0..7,16..23}; lanes 16-31 hold row M=lane-16, K in {8..15,24..31}.
// Each half is a contiguous 16B run in row-major LDS -> 2x ds_load_b128.
__device__ __forceinline__ v16bf load_frag_a(const unsigned short* base, int ldk,
                                             int mtile, int k0, int lane) {
  int row = mtile * 16 + (lane & 15);
  int kd = (lane < 16) ? 0 : 4;                       // dword offset in run
  const unsigned* rp = (const unsigned*)(base + row * ldk) + (k0 >> 1);
  FragU f;
  f.q[0] = *(const uint4*)(rp + kd);                  // K pairs {kb .. kb+7}
  f.q[1] = *(const uint4*)(rp + 8 + kd);              // K pairs {16+kb .. 16+kb+7}
  return f.v;
}

// B-matrix 32x16 bf16 fragment; LDS stores B as [n][k] (row of W = column of B).
// Lanes 0-15: N=lane, K in {0..7,16..23}; lanes 16-31: N=lane-16, K in {8..15,24..31}.
// Two 4-dword runs -> 2x ds_load_b128.
__device__ __forceinline__ v16bf load_frag_b(const unsigned short* base, int ldk,
                                             int ncol0, int k0, int lane) {
  int n = ncol0 + (lane & 15);
  const unsigned* rp = (const unsigned*)(base + n * ldk) + (k0 >> 1)
                       + ((lane < 16) ? 0 : 4);
  FragU f;
  f.q[0] = *(const uint4*)(rp);
  f.q[1] = *(const uint4*)(rp + 8);
  return f.v;
}

__device__ __forceinline__ void grid_barrier(unsigned* bar, unsigned& epoch) {
  __syncthreads();
  if (threadIdx.x == 0) {
    __threadfence();
    atomicAdd(bar, 1u);
    epoch += 1;
    unsigned target = epoch * (unsigned)NWG;
    while (__hip_atomic_load(bar, __ATOMIC_RELAXED, __HIP_MEMORY_SCOPE_AGENT) < target)
      __builtin_amdgcn_s_sleep(1);
    __threadfence();
  }
  __syncthreads();
}

__global__ void init_bar(unsigned* bar) { bar[0] = 0u; }

__global__ __launch_bounds__(NTHREADS)
void decoder_persistent(const float* __restrict__ input, const float* __restrict__ h_enc,
                        const float* __restrict__ W_ih, const float* __restrict__ W_hh,
                        const float* __restrict__ b_ih, const float* __restrict__ b_hh,
                        const float* __restrict__ W_out, const float* __restrict__ b_out,
                        const int* __restrict__ mask0, const int* __restrict__ mask1,
                        const int* __restrict__ skip_p,
                        float* __restrict__ out, float* __restrict__ hbuf,
                        float* __restrict__ gbuf, unsigned* __restrict__ bar) {
  extern __shared__ char smem[];
  const int wg = blockIdx.x;
  const int tid = (int)threadIdx.x;
  const int lane = tid & 31;
  const int wave = tid >> 5;
  const int skip = skip_p[0];
  unsigned epoch = 0;

  // Cooperative copy h_enc -> ping slot 0.
  for (int idx = wg * NTHREADS + tid; idx < B_ * H_; idx += NWG * NTHREADS)
    hbuf[idx] = h_enc[idx];

  if (wg < NGATE_WG) {
    unsigned short* hidA = (unsigned short*)smem;             // 64KB  [128][256] bf16
    unsigned short* xS   = (unsigned short*)(smem + 65536);   // 16KB  [128][64]  bf16
    unsigned short* WhhS = (unsigned short*)(smem + 81920);   // 48KB  [96][256]  bf16
    unsigned short* WihS = (unsigned short*)(smem + 131072);  // 12KB  [96][64]   bf16
    float* gRZ = (float*)(smem + 143360);                     // 32KB  [64][128]
    float* gNI = (float*)(smem + 176128);                     // 16KB  [32][128]
    float* gNH = (float*)(smem + 192512);                     // 16KB  [32][128]
    float* biasRZ = (float*)(smem + 208896);                  // 64 f
    float* biasNI = biasRZ + 64;                              // 32 f
    float* biasNH = biasNI + 32;                              // 32 f

    // Stage weight slices (once) as bf16 in LDS, packed pair stores.
    for (int idx = tid; idx < 96 * (H_ / 2); idx += NTHREADS) {
      int c = idx / (H_ / 2), k2 = (idx - c * (H_ / 2)) * 2;
      int grow = (c >> 5) * H_ + wg * 32 + (c & 31);
      const float* wr = &W_hh[(size_t)grow * H_ + k2];
      *(unsigned*)&WhhS[c * H_ + k2] = pack2bf(wr[0], wr[1]);
    }
    for (int idx = tid; idx < 96 * (X_ / 2); idx += NTHREADS) {
      int c = idx / (X_ / 2), k2 = (idx - c * (X_ / 2)) * 2;
      int grow = (c >> 5) * H_ + wg * 32 + (c & 31);
      const float* wr = &W_ih[(size_t)grow * X_ + k2];
      *(unsigned*)&WihS[c * X_ + k2] = pack2bf(wr[0], wr[1]);
    }
    for (int c = tid; c < 64; c += NTHREADS) {
      int grow = (c >> 5) * H_ + wg * 32 + (c & 31);
      biasRZ[c] = b_ih[grow] + b_hh[grow];
    }
    for (int c = tid; c < 32; c += NTHREADS) {
      int grow = 2 * H_ + wg * 32 + c;
      biasNI[c] = b_ih[grow];
      biasNH[c] = b_hh[grow];
    }
    grid_barrier(bar, epoch);   // init sync

    for (int t = 0; t < T_; ++t) {
      const float* hprev = hbuf + (t & 1) * (B_ * H_);
      float* hnext = hbuf + ((t + 1) & 1) * (B_ * H_);
      float a0 = (float)mask0[t];
      float a1 = (float)mask1[t];
      int pg = (t < skip) ? 2 * t : t - skip;
      int gz = (pg < skip) ? 1 : 0;
      int gidx = pg - skip; if (gidx < 0) gidx = 0;

      // Stage x_{t-1} (GO token at t=0), float4 loads -> ds_store_b64.
      for (int idx = tid; idx < B_ * X_ / 4; idx += NTHREADS) {
        int b = idx >> 4, k4 = (idx & 15) << 2;
        float4 xv = make_float4(0.f, 0.f, 0.f, 0.f);
        if (t > 0) xv = *(const float4*)&input[((size_t)b * T_ + (t - 1)) * X_ + k4];
        *(uint2*)&xS[b * X_ + k4] = make_uint2(pack2bf(xv.x, xv.y), pack2bf(xv.z, xv.w));
        __builtin_prefetch(&input[((size_t)b * T_ + t) * X_ + k4], 0, 1);
      }
      // Stage hidden = h_prev*a0 + skip_g*a1 as bf16 A-matrix.
      for (int idx = tid; idx < B_ * H_ / 4; idx += NTHREADS) {
        int b = idx >> 6, k4 = (idx & 63) << 2;
        float4 hp = *(const float4*)&hprev[b * H_ + k4];
        float4 sg = make_float4(0.f, 0.f, 0.f, 0.f);
        if (!gz) sg = *(const float4*)&gbuf[((size_t)gidx * B_ + b) * H_ + k4];
        float v0 = hp.x * a0 + sg.x * a1, v1 = hp.y * a0 + sg.y * a1;
        float v2 = hp.z * a0 + sg.z * a1, v3 = hp.w * a0 + sg.w * a1;
        *(uint2*)&hidA[b * H_ + k4] = make_uint2(pack2bf(v0, v1), pack2bf(v2, v3));
      }
      __syncthreads();

      const int mtile = wave;  // 8 waves <-> 8 M-tiles of B=128
      v8f acc[6] = {};         // gh accumulators (ct 0..3: r/z combined; 4,5: hn)
      v8f accNI[2] = {};       // gi accumulators for the n gate (inn)
#pragma unroll
      for (int kt = 0; kt < 8; ++kt) {
        v16bf a = load_frag_a(hidA, H_, mtile, kt * 32, lane);
#pragma unroll
        for (int ct = 0; ct < 6; ++ct) {
          v16bf bm = load_frag_b(WhhS, H_, ct * 16, kt * 32, lane);
          acc[ct] = __builtin_amdgcn_wmma_f32_16x16x32_bf16(false, a, false, bm,
                                                            (short)0, acc[ct], false, false);
        }
      }
#pragma unroll
      for (int kt = 0; kt < 2; ++kt) {
        v16bf a = load_frag_a(xS, X_, mtile, kt * 32, lane);
#pragma unroll
        for (int ct = 0; ct < 4; ++ct) {
          v16bf bm = load_frag_b(WihS, X_, ct * 16, kt * 32, lane);
          acc[ct] = __builtin_amdgcn_wmma_f32_16x16x32_bf16(false, a, false, bm,
                                                            (short)0, acc[ct], false, false);
        }
#pragma unroll
        for (int ct = 4; ct < 6; ++ct) {
          v16bf bm = load_frag_b(WihS, X_, ct * 16, kt * 32, lane);
          accNI[ct - 4] = __builtin_amdgcn_wmma_f32_16x16x32_bf16(false, a, false, bm,
                                                                  (short)0, accNI[ct - 4],
                                                                  false, false);
        }
      }
      // C layout: lane N = lane&15; VGPR j -> M = mtile*16 + j + (lane<16 ? 0 : 8).
      // Rows j are contiguous in the [col][row] planes -> 2x ds_store_b128 each.
      {
        int n = lane & 15;
        int m0r = mtile * 16 + ((lane < 16) ? 0 : 8);
#pragma unroll
        for (int ct = 0; ct < 4; ++ct) {
          int col = ct * 16 + n;
          float bb = biasRZ[col];
          float* dst = &gRZ[col * B_ + m0r];
          *(float4*)dst = make_float4(acc[ct][0] + bb, acc[ct][1] + bb,
                                      acc[ct][2] + bb, acc[ct][3] + bb);
          *(float4*)(dst + 4) = make_float4(acc[ct][4] + bb, acc[ct][5] + bb,
                                            acc[ct][6] + bb, acc[ct][7] + bb);
        }
#pragma unroll
        for (int ct = 4; ct < 6; ++ct) {
          int nc = (ct - 4) * 16 + n;
          float bh = biasNH[nc], bi = biasNI[nc];
          float* dh = &gNH[nc * B_ + m0r];
          float* di = &gNI[nc * B_ + m0r];
          v8f ah = acc[ct], ai = accNI[ct - 4];
          *(float4*)dh = make_float4(ah[0] + bh, ah[1] + bh, ah[2] + bh, ah[3] + bh);
          *(float4*)(dh + 4) = make_float4(ah[4] + bh, ah[5] + bh, ah[6] + bh, ah[7] + bh);
          *(float4*)di = make_float4(ai[0] + bi, ai[1] + bi, ai[2] + bi, ai[3] + bi);
          *(float4*)(di + 4) = make_float4(ai[4] + bi, ai[5] + bi, ai[6] + bi, ai[7] + bi);
        }
      }
      __syncthreads();

      // Elementwise gates + state update for this WG's 32 H-columns.
      for (int idx = tid; idx < 32 * B_; idx += NTHREADS) {
        int hl = idx >> 7, b = idx & 127;
        float r  = fsigmoid(gRZ[hl * B_ + b]);
        float z  = fsigmoid(gRZ[(32 + hl) * B_ + b]);
        float nn = ftanh_fast(gNI[hl * B_ + b] + r * gNH[hl * B_ + b]);
        int hc = wg * 32 + hl;
        float hp = hprev[b * H_ + hc];
        float sg = gz ? 0.0f : gbuf[((size_t)gidx * B_ + b) * H_ + hc];
        float hid = hp * a0 + sg * a1;
        float hnew = (1.0f - z) * nn + z * hid;
        hnext[b * H_ + hc] = hnew;
        gbuf[((size_t)t * B_ + b) * H_ + hc] = hnew;   // buf[t] = h_new
      }
      grid_barrier(bar, epoch);  // A: h_new/buf[t] visible to out-WGs
      grid_barrier(bar, epoch);  // B: out-WGs done reading h_prev
    }
  } else {
    // ---- output projection workgroups ----
    const int ow = wg - NGATE_WG;
    unsigned short* hA  = (unsigned short*)smem;              // 64KB [128][256] bf16
    unsigned short* WoS = (unsigned short*)(smem + 65536);    // 8KB  [16][256]  bf16
    float* bO = (float*)(smem + 65536 + 8192);                // 16 f

    for (int idx = tid; idx < 16 * (H_ / 2); idx += NTHREADS) {
      int c = idx >> 7, k2 = (idx & 127) * 2;
      const float* wr = &W_out[(size_t)(ow * 16 + c) * H_ + k2];
      *(unsigned*)&WoS[c * H_ + k2] = pack2bf(wr[0], wr[1]);
    }
    if (tid < 16) bO[tid] = b_out[ow * 16 + tid];
    grid_barrier(bar, epoch);   // init sync (includes __syncthreads)

    // W_out is constant: keep its 8 B-fragments resident in VGPRs for all steps.
    v16bf wof[8];
#pragma unroll
    for (int kt = 0; kt < 8; ++kt) wof[kt] = load_frag_b(WoS, H_, 0, kt * 32, lane);
    float bias = bO[lane & 15];

    for (int t = 0; t < T_; ++t) {
      const float* hprev = hbuf + (t & 1) * (B_ * H_);
      int pp = (t < skip) ? 2 * t + 1 : t - skip;
      int pz = (pp < skip) ? 1 : 0;
      int pidx = pp - skip; if (pidx < 0) pidx = 0;

      grid_barrier(bar, epoch); // A: wait for buf[t] (needed when pidx==t)

      for (int idx = tid; idx < B_ * H_ / 4; idx += NTHREADS) {
        int b = idx >> 6, k4 = (idx & 63) << 2;
        float4 hp = *(const float4*)&hprev[b * H_ + k4];
        float4 sp = make_float4(0.f, 0.f, 0.f, 0.f);
        if (!pz) sp = *(const float4*)&gbuf[((size_t)pidx * B_ + b) * H_ + k4];
        *(uint2*)&hA[b * H_ + k4] = make_uint2(pack2bf(hp.x + sp.x, hp.y + sp.y),
                                               pack2bf(hp.z + sp.z, hp.w + sp.w));
      }
      __syncthreads();

      const int mtile = wave;
      v8f acc = {};
#pragma unroll
      for (int kt = 0; kt < 8; ++kt) {
        v16bf a = load_frag_a(hA, H_, mtile, kt * 32, lane);
        acc = __builtin_amdgcn_wmma_f32_16x16x32_bf16(false, a, false, wof[kt],
                                                      (short)0, acc, false, false);
      }
      int n = lane & 15;
      int m0r = mtile * 16 + ((lane < 16) ? 0 : 8);
      int ocol = ow * 16 + n;
#pragma unroll
      for (int j = 0; j < 8; ++j)
        out[((size_t)(m0r + j) * T_ + t) * X_ + ocol] = acc[j] + bias;

      grid_barrier(bar, epoch); // B: done reading h_prev / hA safe to rewrite
    }
  }
}

extern "C" void kernel_launch(void* const* d_in, const int* in_sizes, int n_in,
                              void* d_out, int out_size, void* d_ws, size_t ws_size,
                              hipStream_t stream) {
  (void)in_sizes; (void)n_in; (void)out_size; (void)ws_size;
  const float* input = (const float*)d_in[0];
  const float* h_enc = (const float*)d_in[1];
  const float* W_ih  = (const float*)d_in[2];
  const float* W_hh  = (const float*)d_in[3];
  const float* b_ih  = (const float*)d_in[4];
  const float* b_hh  = (const float*)d_in[5];
  const float* W_out = (const float*)d_in[6];
  const float* b_out = (const float*)d_in[7];
  const int* mask0 = (const int*)d_in[8];
  const int* mask1 = (const int*)d_in[9];
  const int* skipp = (const int*)d_in[10];
  float* out = (float*)d_out;

  char* ws = (char*)d_ws;
  unsigned* bar = (unsigned*)ws;                               // 256B region
  float* hbuf = (float*)(ws + 256);                            // 2*B*H f32 ping-pong
  float* gbuf = (float*)(ws + 256 + 2 * B_ * H_ * sizeof(float)); // T*B*H f32 history

  hipLaunchKernelGGL(init_bar, dim3(1), dim3(1), 0, stream, bar);
  size_t smem = 210 * 1024;   // dynamic LDS (gate WGs use ~204.5KB; <=320KB/WGP)
  hipLaunchKernelGGL(decoder_persistent, dim3(NWG), dim3(NTHREADS), smem, stream,
                     input, h_enc, W_ih, W_hh, b_ih, b_hh, W_out, b_out,
                     mask0, mask1, skipp, out, hbuf, gbuf, bar);
}
